// GNN1_6528350290209
// MI455X (gfx1250) — compile-verified, matrix operand order checked by reading, not run on hardware
//
#include <hip/hip_runtime.h>
#include <hip/hip_bf16.h>

// ---------------------------------------------------------------------------
// Types for CDNA5 WMMA (wave32): v_wmma_f32_16x16x32_bf16
// ---------------------------------------------------------------------------
typedef __attribute__((ext_vector_type(16))) __bf16 v16bf;
typedef __attribute__((ext_vector_type(8)))  float  v8f;

union Frag {
    v16bf v;
    uint4 q[2];   // 32 bytes
};

__device__ __forceinline__ float bf2f(unsigned short b) {
    return __uint_as_float(((unsigned int)b) << 16);
}
__device__ __forceinline__ unsigned short f2bf(float f) {
    unsigned int u = __float_as_uint(f);
    unsigned int r = u + 0x7FFFu + ((u >> 16) & 1u);   // round-to-nearest-even
    return (unsigned short)(r >> 16);
}

// ---------------------------------------------------------------------------
// fp32 -> bf16 elementwise convert
// ---------------------------------------------------------------------------
__global__ void cvt_f32_bf16(const float* __restrict__ in,
                             unsigned short* __restrict__ out, long long n) {
    long long t = (long long)blockIdx.x * blockDim.x + threadIdx.x;
    if (t < n) out[t] = f2bf(in[t]);
}

// ---------------------------------------------------------------------------
// Pack a row-major fp32 weight [Ksrc x N] into WMMA B fragments (bf16).
// Fragment = 32x16 (KxN) tile laid out exactly as the ISA B-operand layout:
//   lane l holds column n = l&15;  element e (of v16bf):
//   K = e + (e>=8 ? 8 : 0) + (l>=16 ? 8 : 0)      (packed-pair layout, 7.12.2)
// Fragment index = (ktOff + ktl) * (N/16) + nt, 512 bf16 per fragment.
// ---------------------------------------------------------------------------
__global__ void pack_w(const float* __restrict__ W, unsigned short* __restrict__ packed,
                       int N, int Ksrc, int ktOff) {
    int t = blockIdx.x * blockDim.x + threadIdx.x;
    if (t >= Ksrc * N) return;
    int e    = t & 15;
    int lane = (t >> 4) & 31;
    int frag = t >> 9;
    int nFragsN = N >> 4;
    int ktl = frag / nFragsN;
    int nt  = frag - ktl * nFragsN;
    int Kin = e + ((e >= 8) ? 8 : 0) + ((lane >= 16) ? 8 : 0);
    int col = (nt << 4) + (lane & 15);
    int srcK = (ktl << 5) + Kin;
    float v = W[(size_t)srcK * N + col];
    packed[((size_t)(ktOff + ktl) * nFragsN + nt) * 512 + lane * 16 + e] = f2bf(v);
}

// ---------------------------------------------------------------------------
// Edge scatter-add: agg[dst[e], :] += h[src[e], :]   (float4 granularity,
// global_atomic_add_f32; working set is L2-resident on MI455X's 192MB L2)
// ---------------------------------------------------------------------------
__global__ void scatter_f32(const long long* __restrict__ src,
                            const long long* __restrict__ dst,
                            const float* __restrict__ h,
                            float* __restrict__ agg,
                            int d, int shift, long long nWork) {
    long long t = (long long)blockIdx.x * blockDim.x + threadIdx.x;
    if (t >= nWork) return;
    long long e = t >> shift;
    int c = (int)(t & (((long long)1 << shift) - 1)) << 2;
    long long s = src[e], dd = dst[e];
    const float4 v = *(const float4*)(h + s * d + c);
    float* ap = agg + dd * d + c;
    atomicAdd(ap + 0, v.x);
    atomicAdd(ap + 1, v.y);
    atomicAdd(ap + 2, v.z);
    atomicAdd(ap + 3, v.w);
}

__global__ void scatter_bf16(const long long* __restrict__ src,
                             const long long* __restrict__ dst,
                             const unsigned short* __restrict__ h,
                             float* __restrict__ agg,
                             int d, int shift, long long nWork) {
    long long t = (long long)blockIdx.x * blockDim.x + threadIdx.x;
    if (t >= nWork) return;
    long long e = t >> shift;
    int c = (int)(t & (((long long)1 << shift) - 1)) << 2;
    long long s = src[e], dd = dst[e];
    const ushort4 v = *(const ushort4*)(h + s * d + c);
    float* ap = agg + dd * d + c;
    atomicAdd(ap + 0, bf2f(v.x));
    atomicAdd(ap + 1, bf2f(v.y));
    atomicAdd(ap + 2, bf2f(v.z));
    atomicAdd(ap + 3, bf2f(v.w));
}

// ---------------------------------------------------------------------------
// Fused GraphConv GEMM:
//   out[i,:] = act( [A0_row(i) | A1_row(i)] @ Bpacked + bias )
// A0/A1 are row-major bf16 activations (A1 may be null, d1=0).
// Bpacked: pre-packed fragments (see pack_w), K = d0+d1, Nout = N.
//
// Block = 256 threads = 8 wave32s; block tile 256(M) x 64(N); each wave
// computes a 32x64 strip: 2 A fragments x 4 B fragments = 8 WMMAs per
// K-step, with 8 v8f accumulators.
//
// B fragments are staged through a double-buffered 8KB LDS tile (the 4
// fragments a block needs per K-step are 4KB contiguous in the packed
// layout): the global load of tile kt+1 is issued before the WMMAs of kt.
// All 4 B fragments are pulled into distinct register octets so the
// backend can clause the ds_load_b128s and overlap s_wait_dscnt with WMMAs.
// ---------------------------------------------------------------------------
__global__ __launch_bounds__(256)
void gemm_wmma_bf16(const unsigned short* __restrict__ A0, int d0,
                    const unsigned short* __restrict__ A1, int d1,
                    const unsigned short* __restrict__ Bp,
                    const float* __restrict__ bias,
                    unsigned short* __restrict__ outB, int ldOutB,
                    float* __restrict__ outF, int ldOutF,
                    int M, int N, int relu) {
    __shared__ __align__(16) unsigned short Bs[2][4 * 512];   // 2 x 4KB

    const int tid     = threadIdx.x;
    const int lane    = tid & 31;
    const int wave    = tid >> 5;
    const int laneGrp = lane >> 4;        // 0 or 1 (selects K-halves in A frag)
    const int lmod    = lane & 15;
    const int mBase   = blockIdx.x * 256 + wave * 32;
    const int nBase   = blockIdx.y * 64;
    const int K       = d0 + d1;
    const int ktTotal = K >> 5;
    const int nFragsN = N >> 4;
    const int ntB     = nBase >> 4;

    // A-fragment rows for this lane (clamped so WMMA runs with EXEC all-1s)
    int rowA0 = mBase + lmod;
    int rowA1 = mBase + 16 + lmod;
    if (rowA0 > M - 1) rowA0 = M - 1;
    if (rowA1 > M - 1) rowA1 = M - 1;

    v8f acc[2][4];
#pragma unroll
    for (int r = 0; r < 2; ++r)
#pragma unroll
        for (int i = 0; i < 4; ++i)
            acc[r][i] = (v8f){0.f, 0.f, 0.f, 0.f, 0.f, 0.f, 0.f, 0.f};

    // ---- stage B tile for kt=0 ----
    {
        const uint4* gsrc = (const uint4*)(Bp + (size_t)ntB * 512);
        ((uint4*)Bs[0])[tid] = gsrc[tid];
    }
    __syncthreads();

    for (int kt = 0; kt < ktTotal; ++kt) {
        const int cur = kt & 1;
        const bool hasNext = (kt + 1) < ktTotal;

        // Issue next B tile's global load early (completes during WMMAs)
        uint4 nxtData;
        if (hasNext) {
            const uint4* gsrc =
                (const uint4*)(Bp + ((size_t)(kt + 1) * nFragsN + ntB) * 512);
            nxtData = gsrc[tid];
        }

        // ---- A fragments (row-major, two contiguous 16B chunks each) ----
        const int kk = kt << 5;
        const unsigned short* Ap;
        int kloc, lda;
        if (kk < d0) { Ap = A0; kloc = kk;      lda = d0; }
        else         { Ap = A1; kloc = kk - d0; lda = d1; }
        Frag a0, a1;
        const unsigned short* arow0 = Ap + (size_t)rowA0 * lda + kloc + (laneGrp << 3);
        const unsigned short* arow1 = Ap + (size_t)rowA1 * lda + kloc + (laneGrp << 3);
        a0.q[0] = *(const uint4*)(arow0);
        a0.q[1] = *(const uint4*)(arow0 + 16);
        a1.q[0] = *(const uint4*)(arow1);
        a1.q[1] = *(const uint4*)(arow1 + 16);
        __builtin_prefetch(arow0 + 32, 0, 3);   // stream next K-chunk of A
        __builtin_prefetch(arow1 + 32, 0, 3);

        // ---- pull all 4 B fragments into registers (clauseable ds loads) ----
        Frag b[4];
#pragma unroll
        for (int nf = 0; nf < 4; ++nf) {
            const uint4* ls = (const uint4*)(&Bs[cur][nf * 512 + lane * 16]);
            b[nf].q[0] = ls[0];
            b[nf].q[1] = ls[1];
        }

        // ---- 8 WMMAs ----
#pragma unroll
        for (int nf = 0; nf < 4; ++nf) {
            acc[0][nf] = __builtin_amdgcn_wmma_f32_16x16x32_bf16(
                false, a0.v, false, b[nf].v, (short)0, acc[0][nf], false, false);
            acc[1][nf] = __builtin_amdgcn_wmma_f32_16x16x32_bf16(
                false, a1.v, false, b[nf].v, (short)0, acc[1][nf], false, false);
        }

        // ---- commit next tile to the other LDS buffer ----
        if (hasNext) ((uint4*)Bs[cur ^ 1])[tid] = nxtData;
        __syncthreads();
    }

    // Epilogue: bias + optional ReLU; C layout: vgpr v, lane l ->
    //   row = rowBase + (l>=16 ? 8 : 0) + v, col = nBase + 16*nf + (l&15)
    float biasv[4];
#pragma unroll
    for (int nf = 0; nf < 4; ++nf)
        biasv[nf] = bias[nBase + nf * 16 + lmod];

#pragma unroll
    for (int r = 0; r < 2; ++r) {
        const int row0 = mBase + r * 16 + (laneGrp << 3);
#pragma unroll
        for (int v = 0; v < 8; ++v) {
            const int row = row0 + v;
            if (row >= M) continue;
#pragma unroll
            for (int nf = 0; nf < 4; ++nf) {
                float val = acc[r][nf][v] + biasv[nf];
                if (relu) val = fmaxf(val, 0.f);
                const int col = nBase + nf * 16 + lmod;
                if (outB) outB[(size_t)row * ldOutB + col] = f2bf(val);
                if (outF) outF[(size_t)row * ldOutF + col] = val;
            }
        }
    }
}

// ---------------------------------------------------------------------------
// Host-side orchestration
// ---------------------------------------------------------------------------
static inline unsigned int blocks_for(long long n, int bs) {
    return (unsigned int)((n + bs - 1) / bs);
}

extern "C" void kernel_launch(void* const* d_in, const int* in_sizes, int n_in,
                              void* d_out, int out_size, void* d_ws, size_t ws_size,
                              hipStream_t stream) {
    const float* x       = (const float*)d_in[0];
    const float* W1_rel  = (const float*)d_in[1];
    const float* b1      = (const float*)d_in[2];
    const float* W1_root = (const float*)d_in[3];
    const float* W2_rel  = (const float*)d_in[4];
    const float* b2      = (const float*)d_in[5];
    const float* W2_root = (const float*)d_in[6];
    const float* W3_rel  = (const float*)d_in[7];
    const float* b3      = (const float*)d_in[8];
    const float* W3_root = (const float*)d_in[9];
    const float* W_lin   = (const float*)d_in[10];
    const float* b_lin   = (const float*)d_in[11];
    const long long* ei  = (const long long*)d_in[12];

    const int DIN = 128, DH = 512, DOUT = 64;
    const int N  = in_sizes[0] / DIN;         // 50000
    const long long E = in_sizes[12] / 2;     // 1.6M
    const long long* eSrc = ei;
    const long long* eDst = ei + E;

    // ---- workspace layout ----
    char* ws = (char*)d_ws;
    float*          agg  = (float*)ws;                      ws += (size_t)N * DH * 4;
    unsigned short* aggb = (unsigned short*)ws;             ws += (size_t)N * DH * 2;
    unsigned short* hA   = (unsigned short*)ws;             ws += (size_t)N * DH * 2;
    unsigned short* hB   = (unsigned short*)ws;             ws += (size_t)N * DH * 2;
    unsigned short* w1p  = (unsigned short*)ws;             ws += (size_t)8  * 32 * 512 * 2;  // K=256
    unsigned short* w2p  = (unsigned short*)ws;             ws += (size_t)32 * 32 * 512 * 2;  // K=1024
    unsigned short* w3p  = (unsigned short*)ws;             ws += (size_t)32 * 32 * 512 * 2;  // K=1024
    unsigned short* wlp  = (unsigned short*)ws;             ws += (size_t)16 * 4  * 512 * 2;  // K=512,N=64

    float* logits = (float*)d_out;                    // [N, 64]
    float* embed  = (float*)d_out + (size_t)N * DOUT; // [N, 512]

    // ---- pack all weights into WMMA B-fragment layout (cheap, every call) ----
    pack_w<<<blocks_for((long long)DIN * DH, 256), 256, 0, stream>>>(W1_rel,  w1p, DH, DIN, 0);
    pack_w<<<blocks_for((long long)DIN * DH, 256), 256, 0, stream>>>(W1_root, w1p, DH, DIN, DIN / 32);
    pack_w<<<blocks_for((long long)DH  * DH, 256), 256, 0, stream>>>(W2_rel,  w2p, DH, DH, 0);
    pack_w<<<blocks_for((long long)DH  * DH, 256), 256, 0, stream>>>(W2_root, w2p, DH, DH, DH / 32);
    pack_w<<<blocks_for((long long)DH  * DH, 256), 256, 0, stream>>>(W3_rel,  w3p, DH, DH, 0);
    pack_w<<<blocks_for((long long)DH  * DH, 256), 256, 0, stream>>>(W3_root, w3p, DH, DH, DH / 32);
    pack_w<<<blocks_for((long long)DH * DOUT, 256), 256, 0, stream>>>(W_lin,  wlp, DOUT, DH, 0);

    const dim3 blk(256);
    const unsigned int mTiles = (unsigned int)((N + 255) / 256);

    // ================= Layer 1 (d = 128) =================
    cvt_f32_bf16<<<blocks_for((long long)N * DIN, 256), blk, 0, stream>>>(x, hA, (long long)N * DIN);
    hipMemsetAsync(agg, 0, (size_t)N * DIN * 4, stream);
    {
        long long nWork = E << 5;  // d/4 = 32 float4 per edge
        scatter_f32<<<blocks_for(nWork, 256), blk, 0, stream>>>(eSrc, eDst, x, agg, DIN, 5, nWork);
    }
    cvt_f32_bf16<<<blocks_for((long long)N * DIN, 256), blk, 0, stream>>>(agg, aggb, (long long)N * DIN);
    gemm_wmma_bf16<<<dim3(mTiles, DH / 64), blk, 0, stream>>>(
        aggb, DIN, hA, DIN, w1p, b1, hB, DH, nullptr, 0, N, DH, 1);

    // ================= Layer 2 (d = 512) =================
    hipMemsetAsync(agg, 0, (size_t)N * DH * 4, stream);
    {
        long long nWork = E << 7;  // d/4 = 128 float4 per edge
        scatter_bf16<<<blocks_for(nWork, 256), blk, 0, stream>>>(eSrc, eDst, hB, agg, DH, 7, nWork);
    }
    cvt_f32_bf16<<<blocks_for((long long)N * DH, 256), blk, 0, stream>>>(agg, aggb, (long long)N * DH);
    gemm_wmma_bf16<<<dim3(mTiles, DH / 64), blk, 0, stream>>>(
        aggb, DH, hB, DH, w2p, b2, hA, DH, nullptr, 0, N, DH, 1);

    // ================= Layer 3 (d = 512), dual-write fp32 node_embed =========
    hipMemsetAsync(agg, 0, (size_t)N * DH * 4, stream);
    {
        long long nWork = E << 7;
        scatter_bf16<<<blocks_for(nWork, 256), blk, 0, stream>>>(eSrc, eDst, hA, agg, DH, 7, nWork);
    }
    cvt_f32_bf16<<<blocks_for((long long)N * DH, 256), blk, 0, stream>>>(agg, aggb, (long long)N * DH);
    gemm_wmma_bf16<<<dim3(mTiles, DH / 64), blk, 0, stream>>>(
        aggb, DH, hA, DH, w3p, b3, hB, DH, embed, DH, N, DH, 1);

    // ================= Linear head: logits = h3 @ W_lin + b_lin ==============
    gemm_wmma_bf16<<<dim3(mTiles, DOUT / 64), blk, 0, stream>>>(
        hB, DH, nullptr, 0, wlp, b_lin, nullptr, 0, logits, DOUT, N, DOUT, 0);
}
